// LAMModel_70205535420692
// MI455X (gfx1250) — matrix operand order repaired and checked by reading, not compiled, not run on hardware
//
#include <hip/hip_runtime.h>
#include <math.h>

// ---------------------------------------------------------------------------
// Model dims (fixed by the reference)
// ---------------------------------------------------------------------------
#define DD   384
#define HH   12
#define DKK  32
#define LL   6
#define FF_  1536
#define BB   16
#define SS   512
#define TOK  (BB * SS)          // 8192 rows
#define NDH  (TOK * DD)         // 3,145,728 elements

typedef __attribute__((ext_vector_type(16))) __bf16 v16bf;
typedef __attribute__((ext_vector_type(8)))  __bf16 v8bf;
typedef __attribute__((ext_vector_type(8)))  float  v8f;

// ---------------------------------------------------------------------------
// f32 -> bf16 (RNE) conversion for weights / activations
// ---------------------------------------------------------------------------
__global__ void f32_to_bf16_kernel(const float* __restrict__ s,
                                   __bf16* __restrict__ d, int n) {
  int i = blockIdx.x * blockDim.x + threadIdx.x;
  if (i < n) d[i] = (__bf16)s[i];
}

// ---------------------------------------------------------------------------
// WMMA GEMM:  C[M,N] = act( A[M,K](bf16) * B[K,N](bf16) + bias )
// Block tile 128x64, K-step 32, 8 waves; each wave owns a 32x32 patch made of
// four v_wmma_f32_16x16x32_bf16 tiles.  act: 0=none 1=silu 2=gelu 3=sigmoid.
// A tile is staged global->LDS with CDNA5 async copies (ASYNCcnt);
// B tile is transposed through VGPRs so fragments are contiguous ds_load_b128.
// ---------------------------------------------------------------------------
#define TM 128
#define TN 64
#define TK 32

__global__ __launch_bounds__(256) void gemm_bf16_wmma(
    const __bf16* __restrict__ A, const __bf16* __restrict__ Bw,
    const float* __restrict__ bias,
    float* __restrict__ Cf, __bf16* __restrict__ Cbf,
    int M, int N, int K, int act)
{
  // +8 bf16 pad keeps rows 16B aligned and staggers LDS banks
  __shared__ __bf16 As[TM][TK + 8];
  __shared__ __bf16 Bs[TN][TK + 8];   // stored transposed: [n][k]

  const int tid  = threadIdx.x;
  const int lane = tid & 31;
  const int w    = tid >> 5;
  const int l    = lane & 15;   // N / M row-within-tile index
  const int hi   = lane >> 4;   // half-wave select
  const int wr   = w & 3;       // wave row (M), 4 waves
  const int wc   = w >> 2;      // wave col (N), 2 waves
  const int bm   = blockIdx.y * TM;
  const int bn   = blockIdx.x * TN;

  v8f acc[2][2];
#pragma unroll
  for (int mt = 0; mt < 2; ++mt)
#pragma unroll
    for (int nt = 0; nt < 2; ++nt)
#pragma unroll
      for (int r = 0; r < 8; ++r) acc[mt][nt][r] = 0.0f;

  const int arow = tid >> 1;          // 0..127
  const int acol = (tid & 1) * 16;    // 0 / 16 (16 bf16 = two b128)
  const int brow = tid >> 3;          // 0..31  (k)
  const int bcol = (tid & 7) * 8;     // 0..56  (n, 8 bf16 = one b128)

  // Wave-relative LDS byte offset of this thread's A-tile slot (flat shared
  // pointers carry the AS3 offset in their low 32 bits).
  const unsigned ldsA = (unsigned)(size_t)&As[arow][acol];

  for (int k0 = 0; k0 < K; k0 += TK) {
    // B chunk: global -> VGPR (will be LDS-transposed after the barrier)
    uint4 bv = *reinterpret_cast<const uint4*>(
        &Bw[(size_t)(k0 + brow) * N + bn + bcol]);
    if (k0 + TK < K) {  // CDNA5 global_prefetch_b8 for the next K tile
      __builtin_prefetch(&A[(size_t)(bm + arow) * K + k0 + TK + acol]);
      __builtin_prefetch(&Bw[(size_t)(k0 + TK + brow) * N + bn + bcol]);
    }
    __syncthreads();   // consumers of the previous tile are done

    // A tile: one async b128 burst per thread, memory -> LDS directly
    // (tracked by ASYNCcnt; no VGPR staging, no ds_store issue slots).
    {
      unsigned long long ga =
          (unsigned long long)(size_t)&A[(size_t)(bm + arow) * K + k0 + acol];
      asm volatile("global_load_async_to_lds_b128 %0, %1, off"
                   :: "v"(ldsA), "v"(ga)
                   : "memory");
    }
    // B tile: scatter-transpose into Bs[n][k]
    const __bf16* bp = reinterpret_cast<const __bf16*>(&bv);
#pragma unroll
    for (int e = 0; e < 8; ++e) Bs[bcol + e][brow] = bp[e];

    // Each wave drains its own async copies, then the barrier makes every
    // wave's LDS writes (async A + ds B) visible block-wide.
    asm volatile("s_wait_asynccnt 0x0" ::: "memory");
    __syncthreads();

    // A fragment (16-bit A 16x32 layout): lane holds row M=l; element e maps
    // to K = 8*hi + e (e<8) and K = 16 + 8*hi + (e-8) (e>=8): two 16B chunks.
    v16bf afrag[2], bfrag[2];
#pragma unroll
    for (int mt = 0; mt < 2; ++mt) {
      int r = wr * 32 + mt * 16 + l;
      v8bf a0 = *reinterpret_cast<const v8bf*>(&As[r][hi * 8]);
      v8bf a1 = *reinterpret_cast<const v8bf*>(&As[r][16 + hi * 8]);
      afrag[mt] = __builtin_shufflevector(a0, a1, 0, 1, 2, 3, 4, 5, 6, 7,
                                          8, 9, 10, 11, 12, 13, 14, 15);
    }
    // B fragment (16-bit B 32x16): lane holds col N=l; K = 8*hi + e (e<8), 16 + 8*hi + (e-8) (e>=8).
#pragma unroll
    for (int nt = 0; nt < 2; ++nt) {
      int c = wc * 32 + nt * 16 + l;
      v8bf b0 = *reinterpret_cast<const v8bf*>(&Bs[c][hi * 8]);
      v8bf b1 = *reinterpret_cast<const v8bf*>(&Bs[c][16 + hi * 8]);
      bfrag[nt] = __builtin_shufflevector(b0, b1, 0, 1, 2, 3, 4, 5, 6, 7,
                                          8, 9, 10, 11, 12, 13, 14, 15);
    }
#pragma unroll
    for (int mt = 0; mt < 2; ++mt)
#pragma unroll
      for (int nt = 0; nt < 2; ++nt)
        acc[mt][nt] = __builtin_amdgcn_wmma_f32_16x16x32_bf16(
            false, afrag[mt], false, bfrag[nt], (short)0, acc[mt][nt],
            false, false);
  }

  // Epilogue: C/D layout -> lane l = N, VGPR r -> M = r + 8*hi
#pragma unroll
  for (int mt = 0; mt < 2; ++mt)
#pragma unroll
    for (int nt = 0; nt < 2; ++nt) {
      int col = bn + wc * 32 + nt * 16 + l;
      float bval = bias ? bias[col] : 0.0f;
#pragma unroll
      for (int r = 0; r < 8; ++r) {
        int row = bm + wr * 32 + mt * 16 + hi * 8 + r;
        float x = acc[mt][nt][r] + bval;
        if (act == 1)      x = x / (1.0f + __expf(-x));                      // silu
        else if (act == 2) x = 0.5f * x * (1.0f + erff(x * 0.70710678f));    // gelu
        else if (act == 3) x = 1.0f / (1.0f + __expf(-x));                   // sigmoid
        size_t idx = (size_t)row * N + col;
        if (Cf)  Cf[idx]  = x;
        if (Cbf) Cbf[idx] = (__bf16)x;
      }
    }
}

// ---------------------------------------------------------------------------
// Embedding sum (tok + pos + type0); LN applied by ln_kernel afterwards
// ---------------------------------------------------------------------------
__global__ __launch_bounds__(128) void embed_kernel(
    const int* __restrict__ ids, const float* __restrict__ tok,
    const float* __restrict__ pos, const float* __restrict__ typ,
    float* __restrict__ h)
{
  int i = blockIdx.x;            // token row (b*S + s)
  int t = threadIdx.x;
  int id = ids[i];
  int s  = i & (SS - 1);
#pragma unroll
  for (int r = 0; r < 3; ++r) {
    int d = t + 128 * r;
    h[(size_t)i * DD + d] =
        tok[(size_t)id * DD + d] + pos[(size_t)s * DD + d] + typ[d];
  }
}

// ---------------------------------------------------------------------------
// Fused (residual +) LayerNorm; writes f32 output and bf16 copy for WMMA.
// One 128-thread block per row of 384.
// ---------------------------------------------------------------------------
__global__ __launch_bounds__(128) void ln_kernel(
    const float* __restrict__ resid, const float* __restrict__ x,
    const float* __restrict__ g, const float* __restrict__ b,
    float* __restrict__ y, __bf16* __restrict__ ybf)
{
  __shared__ float red[128];
  int row = blockIdx.x;
  int t   = threadIdx.x;
  float vals[3];
  float s = 0.0f;
#pragma unroll
  for (int r = 0; r < 3; ++r) {
    int d = t + 128 * r;
    float xv = x[(size_t)row * DD + d];
    if (resid) xv += resid[(size_t)row * DD + d];
    vals[r] = xv;
    s += xv;
  }
  red[t] = s;
  __syncthreads();
  for (int o = 64; o > 0; o >>= 1) {
    if (t < o) red[t] += red[t + o];
    __syncthreads();
  }
  float mean = red[0] * (1.0f / DD);
  __syncthreads();
  float sq = 0.0f;
#pragma unroll
  for (int r = 0; r < 3; ++r) {
    float c = vals[r] - mean;
    sq += c * c;
  }
  red[t] = sq;
  __syncthreads();
  for (int o = 64; o > 0; o >>= 1) {
    if (t < o) red[t] += red[t + o];
    __syncthreads();
  }
  float rstd = rsqrtf(red[0] * (1.0f / DD) + 1e-12f);
#pragma unroll
  for (int r = 0; r < 3; ++r) {
    int d = t + 128 * r;
    float yv = (vals[r] - mean) * rstd * g[d] + b[d];
    y[(size_t)row * DD + d] = yv;
    if (ybf) ybf[(size_t)row * DD + d] = (__bf16)yv;
  }
}

// ---------------------------------------------------------------------------
// beta = sigmoid(x @ Wbeta + bbeta) * mask      (N=12, not WMMA-shaped)
// ---------------------------------------------------------------------------
__global__ void beta_kernel(const float* __restrict__ x,
                            const float* __restrict__ Wb,
                            const float* __restrict__ bb,
                            const int* __restrict__ mask,
                            float* __restrict__ beta)
{
  int t = blockIdx.x * blockDim.x + threadIdx.x;
  if (t >= TOK * HH) return;
  int i = t / HH, h = t % HH;
  float acc = bb[h];
  for (int d = 0; d < DD; ++d)
    acc += x[(size_t)i * DD + d] * Wb[d * HH + h];
  beta[t] = (1.0f / (1.0f + __expf(-acc))) * (float)mask[i];
}

// ---------------------------------------------------------------------------
// Hierarchical delta-rule recurrence.  One wave32 per (b,h) pair; lane j
// holds column j of the fast and slow 32x32 memory states in VGPRs.
// k-normalization fused; k/q broadcast through LDS; wave-reduce for |k|.
// ---------------------------------------------------------------------------
__global__ __launch_bounds__(32) void delta_recurrence(
    const float* __restrict__ q, const float* __restrict__ k,
    const float* __restrict__ v, const float* __restrict__ beta,
    const float* __restrict__ df_logit, const float* __restrict__ ds_logit,
    const float* __restrict__ mix_logit, float* __restrict__ o)
{
  const int bh = blockIdx.x;
  const int b  = bh / HH;
  const int h  = bh % HH;
  const int j  = threadIdx.x;

  const float df  = 1.0f / (1.0f + __expf(-df_logit[h]));
  const float ds  = 1.0f / (1.0f + __expf(-ds_logit[h]));
  const float mix = 1.0f / (1.0f + __expf(-mix_logit[h]));

  float Sf[DKK], Ss[DKK];
#pragma unroll
  for (int t = 0; t < DKK; ++t) { Sf[t] = 0.0f; Ss[t] = 0.0f; }

  __shared__ float sk[DKK];
  __shared__ float sq[DKK];

  for (int s = 0; s < SS; ++s) {
    size_t base = ((size_t)(b * SS + s)) * DD + h * DKK;
    float kj = k[base + j];
    float qj = q[base + j];
    float vj = v[base + j];
    // ||k|| via wave32 xor-reduction
    float n2 = kj * kj;
#pragma unroll
    for (int off = 16; off > 0; off >>= 1) n2 += __shfl_xor(n2, off, 32);
    kj = kj / (sqrtf(n2) + 1e-6f);
    sk[j] = kj;
    sq[j] = qj;
    __syncthreads();

    float bt = beta[(size_t)(b * SS + s) * HH + h];
    float dotf = 0.0f, dots = 0.0f;
#pragma unroll
    for (int t = 0; t < DKK; ++t) {
      dotf += sk[t] * Sf[t];
      dots += sk[t] * Ss[t];
    }
    float errf = (vj - dotf) * bt;
    float errs = (vj - dots) * bt;
    float of = 0.0f, os = 0.0f;
#pragma unroll
    for (int t = 0; t < DKK; ++t) {
      Sf[t] = df * Sf[t] + sk[t] * errf;
      of   += sq[t] * Sf[t];
      Ss[t] = ds * Ss[t] + sk[t] * errs;
      os   += sq[t] * Ss[t];
    }
    o[base + j] = mix * of + (1.0f - mix) * os;
    __syncthreads();
  }
}

// ---------------------------------------------------------------------------
// ogbf = bf16( o * gsig )   (gate already sigmoid'ed by the GEMM epilogue)
// ---------------------------------------------------------------------------
__global__ void gate_mul_kernel(const float* __restrict__ o,
                                const float* __restrict__ gsig,
                                __bf16* __restrict__ ogbf, int n)
{
  int i = blockIdx.x * blockDim.x + threadIdx.x;
  if (i < n) ogbf[i] = (__bf16)(o[i] * gsig[i]);
}

// ---------------------------------------------------------------------------
// Masked mean-pool over S then L2 normalize.  One block per batch row.
// ---------------------------------------------------------------------------
__global__ __launch_bounds__(384) void pool_kernel(
    const float* __restrict__ h, const int* __restrict__ mask,
    float* __restrict__ out)
{
  __shared__ float red[DD];
  int b = blockIdx.x;
  int d = threadIdx.x;
  float acc = 0.0f, msum = 0.0f;
  for (int s = 0; s < SS; ++s) {
    float m = (float)mask[b * SS + s];
    acc  += h[((size_t)(b * SS + s)) * DD + d] * m;
    msum += m;
  }
  float e = acc / fmaxf(msum, 1e-9f);
  red[d] = e * e;
  __syncthreads();
  if (d == 0) {
    float s = 0.0f;
    for (int i = 0; i < DD; ++i) s += red[i];
    red[0] = s;
  }
  __syncthreads();
  out[(size_t)b * DD + d] = e / (sqrtf(red[0]) + 1e-12f);
}

// ---------------------------------------------------------------------------
// Host orchestration
// ---------------------------------------------------------------------------
extern "C" void kernel_launch(void* const* d_in, const int* in_sizes, int n_in,
                              void* d_out, int out_size, void* d_ws, size_t ws_size,
                              hipStream_t stream) {
  (void)in_sizes; (void)n_in; (void)out_size; (void)ws_size;

  const int*   ids      = (const int*)  d_in[0];
  const int*   amask    = (const int*)  d_in[1];
  const float* emb_tok  = (const float*)d_in[2];
  const float* emb_pos  = (const float*)d_in[3];
  const float* emb_type = (const float*)d_in[4];
  const float* emb_ln_g = (const float*)d_in[5];
  const float* emb_ln_b = (const float*)d_in[6];
  const float* Wq    = (const float*)d_in[7];
  const float* bq    = (const float*)d_in[8];
  const float* Wk    = (const float*)d_in[9];
  const float* bk    = (const float*)d_in[10];
  const float* Wv    = (const float*)d_in[11];
  const float* bv    = (const float*)d_in[12];
  const float* Wbeta = (const float*)d_in[13];
  const float* bbeta = (const float*)d_in[14];
  const float* dfp   = (const float*)d_in[15];
  const float* dsp   = (const float*)d_in[16];
  const float* mixp  = (const float*)d_in[17];
  const float* Wg    = (const float*)d_in[18];
  const float* Wo    = (const float*)d_in[19];
  const float* bo    = (const float*)d_in[20];
  const float* aln_g = (const float*)d_in[21];
  const float* aln_b = (const float*)d_in[22];
  const float* Wf1   = (const float*)d_in[23];
  const float* bf1   = (const float*)d_in[24];
  const float* Wf2   = (const float*)d_in[25];
  const float* bf2   = (const float*)d_in[26];
  const float* fln_g = (const float*)d_in[27];
  const float* fln_b = (const float*)d_in[28];

  // workspace bump allocator (256B aligned)
  char*  ws  = (char*)d_ws;
  size_t off = 0;
  auto alloc = [&](size_t bytes) -> void* {
    void* p = ws + off;
    off = (off + bytes + 255) & ~(size_t)255;
    return p;
  };
  float*  h     = (float*) alloc((size_t)NDH * 4);
  __bf16* xbf   = (__bf16*)alloc((size_t)NDH * 2);
  float*  qf    = (float*) alloc((size_t)NDH * 4);
  float*  kf    = (float*) alloc((size_t)NDH * 4);
  float*  vf    = (float*) alloc((size_t)NDH * 4);
  float*  ob    = (float*) alloc((size_t)NDH * 4);
  float*  tmp   = (float*) alloc((size_t)NDH * 4);  // gsig, then Wo/FFN2 output
  float*  betab = (float*) alloc((size_t)TOK * HH * 4);
  __bf16* ogbf  = (__bf16*)alloc((size_t)NDH * 2);
  __bf16* fmid  = (__bf16*)alloc((size_t)TOK * FF_ * 2);
  __bf16* wqb   = (__bf16*)alloc((size_t)LL * DD * DD * 2);
  __bf16* wkb   = (__bf16*)alloc((size_t)LL * DD * DD * 2);
  __bf16* wvb   = (__bf16*)alloc((size_t)LL * DD * DD * 2);
  __bf16* wgb   = (__bf16*)alloc((size_t)LL * DD * DD * 2);
  __bf16* wob   = (__bf16*)alloc((size_t)LL * DD * DD * 2);
  __bf16* w1b   = (__bf16*)alloc((size_t)LL * DD * FF_ * 2);
  __bf16* w2b   = (__bf16*)alloc((size_t)LL * FF_ * DD * 2);

  // --- weight conversion (deterministic, re-done every launch) ---
  const int nDD = LL * DD * DD;    // 884,736
  const int nDF = LL * DD * FF_;   // 3,538,944
  f32_to_bf16_kernel<<<(nDD + 255) / 256, 256, 0, stream>>>(Wq, wqb, nDD);
  f32_to_bf16_kernel<<<(nDD + 255) / 256, 256, 0, stream>>>(Wk, wkb, nDD);
  f32_to_bf16_kernel<<<(nDD + 255) / 256, 256, 0, stream>>>(Wv, wvb, nDD);
  f32_to_bf16_kernel<<<(nDD + 255) / 256, 256, 0, stream>>>(Wg, wgb, nDD);
  f32_to_bf16_kernel<<<(nDD + 255) / 256, 256, 0, stream>>>(Wo, wob, nDD);
  f32_to_bf16_kernel<<<(nDF + 255) / 256, 256, 0, stream>>>(Wf1, w1b, nDF);
  f32_to_bf16_kernel<<<(nDF + 255) / 256, 256, 0, stream>>>(Wf2, w2b, nDF);

  // --- embedding + LN ---
  embed_kernel<<<TOK, 128, 0, stream>>>(ids, emb_tok, emb_pos, emb_type, h);
  ln_kernel<<<TOK, 128, 0, stream>>>(nullptr, h, emb_ln_g, emb_ln_b, h, xbf);

  const dim3 gN384(DD / TN, TOK / TM);   // (6, 64)
  const dim3 gN1536(FF_ / TN, TOK / TM); // (24, 64)

  for (int i = 0; i < LL; ++i) {
    const size_t wDD = (size_t)i * DD * DD;
    const size_t wDF = (size_t)i * DD * FF_;
    // projections
    gemm_bf16_wmma<<<gN384, 256, 0, stream>>>(xbf, wqb + wDD, bq + i * DD,
                                              qf, nullptr, TOK, DD, DD, 1);
    gemm_bf16_wmma<<<gN384, 256, 0, stream>>>(xbf, wkb + wDD, bk + i * DD,
                                              kf, nullptr, TOK, DD, DD, 1);
    gemm_bf16_wmma<<<gN384, 256, 0, stream>>>(xbf, wvb + wDD, bv + i * DD,
                                              vf, nullptr, TOK, DD, DD, 0);
    beta_kernel<<<(TOK * HH + 255) / 256, 256, 0, stream>>>(
        h, Wbeta + (size_t)i * DD * HH, bbeta + i * HH, amask, betab);
    gemm_bf16_wmma<<<gN384, 256, 0, stream>>>(xbf, wgb + wDD, nullptr,
                                              tmp, nullptr, TOK, DD, DD, 3);
    // sequential delta-rule recurrence: one wave per (b,h)
    delta_recurrence<<<BB * HH, 32, 0, stream>>>(
        qf, kf, vf, betab, dfp + i * HH, dsp + i * HH, mixp + i * HH, ob);
    gate_mul_kernel<<<(NDH + 255) / 256, 256, 0, stream>>>(ob, tmp, ogbf, NDH);
    gemm_bf16_wmma<<<gN384, 256, 0, stream>>>(ogbf, wob + wDD, bo + i * DD,
                                              tmp, nullptr, TOK, DD, DD, 0);
    ln_kernel<<<TOK, 128, 0, stream>>>(h, tmp, aln_g + i * DD, aln_b + i * DD,
                                       h, xbf);
    // FFN
    gemm_bf16_wmma<<<gN1536, 256, 0, stream>>>(xbf, w1b + wDF, bf1 + i * FF_,
                                               nullptr, fmid, TOK, FF_, DD, 2);
    gemm_bf16_wmma<<<gN384, 256, 0, stream>>>(fmid, w2b + wDF, bf2 + i * DD,
                                              tmp, nullptr, TOK, DD, FF_, 0);
    ln_kernel<<<TOK, 128, 0, stream>>>(h, tmp, fln_g + i * DD, fln_b + i * DD,
                                       h, xbf);
  }

  pool_kernel<<<BB, DD, 0, stream>>>(h, amask, (float*)d_out);
}
